// VQVAE_ENCODER_61813169324041
// MI455X (gfx1250) — matrix operand order repaired and checked by reading, not compiled, not run on hardware
//
#include <hip/hip_runtime.h>

typedef __attribute__((ext_vector_type(16))) _Float16 v16h;
typedef __attribute__((ext_vector_type(8)))  float    v8f;
typedef unsigned int u32;
typedef __attribute__((ext_vector_type(4)))  u32 u32x4;
typedef __attribute__((ext_vector_type(4)))  int i32x4;
typedef __attribute__((ext_vector_type(8)))  int i32x8;

#define BN_EPS 1e-5f

// ---------------- workspace layout (in floats) ----------------
// stats: layer L at [L*128, L*128+128): sum[c] at +c, sumsq[c] at +64+c
// bn:    layer L at [512+L*128, ...):   scale[c] at +c, shift[c] at +64+c
static const size_t WS_STATS = 0;
static const size_t WS_BN    = 512;
static const size_t WS_Y1    = 1024;                       // (128,128,128,16) NHWC
static const size_t WS_Y2    = WS_Y1 + 33554432ull;        // (128,64,64,32)   NHWC
static const size_t WS_Y3    = WS_Y2 + 16777216ull;        // (128,32,32,64)   NHWC
static const size_t WS_ZPRE  = WS_Y3 + 8388608ull;         // (131072,32)

// ---- Tensor Data Mover: 2-D tile load, global -> LDS (f32 elements) ----
// D# group0: count=1, lds_addr[63:32], global_addr[120:64], type=2 [127:126]
// D# group1: data_size[17:16], tensor_dim0[79:48], tensor_dim1[111:80],
//            tile_dim0[127:112], tile_dim1[143:128], dim0_stride[207:160]
__device__ inline void tdm_load_2d_f32(void* lds_dst, const float* gsrc,
                                       u32 dim0, u32 dim1,
                                       u32 tile0, u32 tile1, u32 stride0)
{
    u32 lds = (u32)(size_t)lds_dst;
    unsigned long long ga = (unsigned long long)(size_t)gsrc;
    u32x4 g0 = {};
    g0[0] = 1u;                                            // count=1, no gather
    g0[1] = lds;
    g0[2] = (u32)(ga & 0xffffffffu);
    g0[3] = (u32)((ga >> 32) & 0x1ffffffu) | (2u << 30);   // type=2 ("image")
    i32x8 g1 = {};
    g1[0] = (int)(2u << 16);                               // data_size=2 -> 4 bytes
    g1[1] = (int)((dim0 & 0xffffu) << 16);
    g1[2] = (int)((dim0 >> 16) | ((dim1 & 0xffffu) << 16));
    g1[3] = (int)((dim1 >> 16) | (tile0 << 16));
    g1[4] = (int)(tile1 & 0xffffu);
    g1[5] = (int)stride0;
    i32x4 z4 = {};
#if __clang_major__ >= 23
    i32x8 z8 = {};
    __builtin_amdgcn_tensor_load_to_lds(g0, g1, z4, z4, z8, 0);
#else
    __builtin_amdgcn_tensor_load_to_lds(g0, g1, z4, z4, 0);
#endif
}

// ---------------------------------------------------------------
__global__ void k_zero_stats(float* __restrict__ s) {
    int t = threadIdx.x;
    s[t] = 0.f;
    s[t + 256] = 0.f;
}

// conv1: Cin=1, 4x4, s2, p1 : x (128,1,256,256) -> y1 NHWC (128,128,128,16)
__global__ __launch_bounds__(256) void k_conv1(
    const float* __restrict__ x, const float* __restrict__ W1,
    const float* __restrict__ b1, float* __restrict__ y1,
    float* __restrict__ stat)
{
    __shared__ float w[256];      // [tap*16 + co]
    __shared__ float ls[16], lss[16];
    int tid = threadIdx.x;
    {   // W1 OIHW (16,1,4,4) -> w[(kh*4+kw)*16 + co]
        int co = tid & 15, tap = tid >> 4;
        w[tap * 16 + co] = W1[co * 16 + tap];
    }
    if (tid < 16) { ls[tid] = 0.f; lss[tid] = 0.f; }
    __syncthreads();

    int m = blockIdx.x * 256 + tid;           // output pixel (n,oh,ow)
    int n = m >> 14, rem = m & 16383, oh = rem >> 7, ow = rem & 127;

    float acc[16];
    #pragma unroll
    for (int c = 0; c < 16; ++c) acc[c] = b1[c];

    #pragma unroll
    for (int kh = 0; kh < 4; ++kh) {
        int ih = oh * 2 - 1 + kh;
        if (ih < 0 || ih >= 256) continue;
        #pragma unroll
        for (int kw = 0; kw < 4; ++kw) {
            int iw = ow * 2 - 1 + kw;
            if (iw < 0 || iw >= 256) continue;
            float xv = x[((size_t)n << 16) + ih * 256 + iw];
            const float* wr = &w[(kh * 4 + kw) * 16];
            #pragma unroll
            for (int c = 0; c < 16; ++c) acc[c] = fmaf(xv, wr[c], acc[c]);
        }
    }
    float4* o = (float4*)&y1[(size_t)m * 16];
    o[0] = make_float4(acc[0], acc[1], acc[2], acc[3]);
    o[1] = make_float4(acc[4], acc[5], acc[6], acc[7]);
    o[2] = make_float4(acc[8], acc[9], acc[10], acc[11]);
    o[3] = make_float4(acc[12], acc[13], acc[14], acc[15]);

    #pragma unroll
    for (int c = 0; c < 16; ++c) {
        atomicAdd(&ls[c], acc[c]);
        atomicAdd(&lss[c], acc[c] * acc[c]);
    }
    __syncthreads();
    if (tid < 16) {
        atomicAdd(&stat[tid], ls[tid]);
        atomicAdd(&stat[64 + tid], lss[tid]);
    }
}

// fold batch stats -> per-channel scale/shift
__global__ void k_bnfold(const float* __restrict__ stat,
                         const float* __restrict__ g, const float* __restrict__ be,
                         float* __restrict__ bn, int C, float inv_count)
{
    int c = threadIdx.x;
    if (c < C) {
        float mean = stat[c] * inv_count;
        float var  = stat[64 + c] * inv_count - mean * mean;
        float sc   = g[c] * rsqrtf(var + BN_EPS);
        bn[c]      = sc;
        bn[64 + c] = be[c] - mean * sc;
    }
}

// -------- implicit-GEMM conv via v_wmma_f32_16x16x32_f16 --------
template<int CIN, int COUT, int HIN, int WIN, int HOUT, int WOUT, int KS, int STR, int PAD>
__global__ __launch_bounds__(256) void k_conv_wmma(
    const float* __restrict__ in, const float* __restrict__ Wt /*OIHW*/,
    const float* __restrict__ bias, const float* __restrict__ bn_in,
    float* __restrict__ out, float* __restrict__ stat)
{
    constexpr int KTOT = CIN * KS * KS;
    constexpr int NT   = COUT / 16;     // 16x16 C-tiles along N
    constexpr int NCH  = KTOT / 32;     // K chunks of 32

    __shared__ _Float16 Bt[COUT * KTOT];        // [n][k], f16
    __shared__ float s_sc[CIN], s_sh[CIN];
    __shared__ float ls[COUT], lss[COUT];

    int tid = threadIdx.x;
    for (int idx = tid; idx < COUT * KTOT; idx += 256) {
        int nco = idx / KTOT, k = idx % KTOT;
        int tap = k / CIN, ci = k % CIN;
        int kh = tap / KS, kw = tap % KS;
        Bt[idx] = (_Float16)Wt[((nco * CIN + ci) * KS + kh) * KS + kw];
    }
    for (int c = tid; c < CIN; c += 256) { s_sc[c] = bn_in[c]; s_sh[c] = bn_in[64 + c]; }
    for (int c = tid; c < COUT; c += 256) { ls[c] = 0.f; lss[c] = 0.f; }
    __syncthreads();

    int lane = tid & 31, wave = tid >> 5, hs = lane >> 4;
    int m_base = blockIdx.x * 128 + wave * 16;
    int m = m_base + (lane & 15);
    int nimg = m / (HOUT * WOUT);
    int rem  = m % (HOUT * WOUT);
    int oh = rem / WOUT, ow = rem % WOUT;

    __builtin_prefetch(&in[(((size_t)nimg * HIN + (oh * STR)) * WIN) * CIN], 0, 0);

    v8f acc[NT];
    #pragma unroll
    for (int t = 0; t < NT; ++t) acc[t] = (v8f){};

    for (int ch = 0; ch < NCH; ++ch) {
        // A fragment: 16-bit A layout -> lane-half hs holds K runs
        // {ch*32 + hs*8 .. +7} and {ch*32 + 16 + hs*8 .. +7}
        v16h a;
        #pragma unroll
        for (int run = 0; run < 2; ++run) {
            int k0  = ch * 32 + run * 16 + hs * 8;
            int tap = k0 / CIN, ci0 = k0 % CIN;   // 8-run stays within one tap (CIN>=16)
            int kh = tap / KS, kw = tap % KS;
            int ih = oh * STR - PAD + kh, iw = ow * STR - PAD + kw;
            float v[8];
            if (ih >= 0 && ih < HIN && iw >= 0 && iw < WIN) {
                const float4* p =
                    (const float4*)&in[(((size_t)nimg * HIN + ih) * WIN + iw) * CIN + ci0];
                float4 q0 = p[0], q1 = p[1];
                v[0] = q0.x; v[1] = q0.y; v[2] = q0.z; v[3] = q0.w;
                v[4] = q1.x; v[5] = q1.y; v[6] = q1.z; v[7] = q1.w;
                #pragma unroll
                for (int j = 0; j < 8; ++j)   // BN + ReLU of input activation
                    v[j] = fmaxf(fmaf(v[j], s_sc[ci0 + j], s_sh[ci0 + j]), 0.f);
            } else {
                #pragma unroll
                for (int j = 0; j < 8; ++j) v[j] = 0.f;  // zero pad (post-activation)
            }
            #pragma unroll
            for (int j = 0; j < 8; ++j) a[run * 8 + j] = (_Float16)v[j];
        }
        // B fragments from LDS + WMMA
        #pragma unroll
        for (int t = 0; t < NT; ++t) {
            int ncol = t * 16 + (lane & 15);
            const _Float16* bp = &Bt[ncol * KTOT + ch * 32 + hs * 8];
            v16h b;
            #pragma unroll
            for (int j = 0; j < 16; ++j) b[j] = bp[(j & 8) + j];
            acc[t] = __builtin_amdgcn_wmma_f32_16x16x32_f16(
                false, a, false, b, (short)0, acc[t], false, false);
        }
    }

    // epilogue: bias, store NHWC, per-channel stats
    #pragma unroll
    for (int t = 0; t < NT; ++t) {
        int chn = t * 16 + (lane & 15);
        float bs = bias[chn];
        float s = 0.f, ss = 0.f;
        #pragma unroll
        for (int r = 0; r < 8; ++r) {
            float v = acc[t][r] + bs;
            int mrow = m_base + r + hs * 8;   // C layout: vgpr r -> M=r (+8 hi half)
            out[(size_t)mrow * COUT + chn] = v;
            s += v; ss += v * v;
        }
        atomicAdd(&ls[chn], s);
        atomicAdd(&lss[chn], ss);
    }
    __syncthreads();
    for (int c = tid; c < COUT; c += 256) {
        atomicAdd(&stat[c], ls[c]);
        atomicAdd(&stat[64 + c], lss[c]);
    }
}

// -------- VQ: TDM-staged codebook + z tile, dists via WMMA --------
__global__ __launch_bounds__(256) void k_vq(
    const float* __restrict__ zpre,   // (M=131072, 32)
    const float* __restrict__ emb,    // (512, 32)
    const float* __restrict__ bn4,
    float* __restrict__ zex,          // NCHW (128,32,32,32)
    float* __restrict__ zqx,          // NCHW
    int* __restrict__ lat)            // (128,32,32)
{
    __shared__ float    Ef32[512 * 32];   // 64 KB, filled by TDM
    __shared__ float    zstage[128 * 32]; // 16 KB, filled by TDM
    __shared__ _Float16 E[512 * 32];      // 32 KB f16 codebook
    __shared__ float en[512];             // ||e_n||^2
    __shared__ float s_sc[32], s_sh[32];
    __shared__ float zn[128];             // ||z_m||^2 per block row
    __shared__ int   best[128];

    int tid = threadIdx.x;
    int lane = tid & 31, wave = tid >> 5, hs = lane >> 4;

    // ---- stage 0: Tensor Data Mover DMA (wave 0 issues, block barriers) ----
    if (wave == 0) {
        tdm_load_2d_f32(Ef32, emb, 32u, 512u, 32u, 512u, 32u);
        tdm_load_2d_f32(zstage, zpre + (size_t)blockIdx.x * 4096, 32u, 128u, 32u, 128u, 32u);
        __builtin_amdgcn_s_wait_tensorcnt(0);
    }
    __syncthreads();   // zstage/Ef32 visible to all waves

    // ---- stage 1: f16 conversion, ||e||^2, BN constants ----
    for (int idx = tid; idx < 512 * 32; idx += 256) E[idx] = (_Float16)Ef32[idx];
    if (tid < 32) { s_sc[tid] = bn4[tid]; s_sh[tid] = bn4[64 + tid]; }
    for (int n = tid; n < 512; n += 256) {
        float s = 0.f;
        #pragma unroll
        for (int c = 0; c < 32; ++c) { float e = Ef32[n * 32 + c]; s += e * e; }
        en[n] = s;
    }
    __syncthreads();

    int m_base = blockIdx.x * 128 + wave * 16;
    int lr = wave * 16 + (lane & 15);          // local row in zstage
    int m = m_base + (lane & 15);
    int nimg = m >> 10, rem = m & 1023, h = rem >> 5, w = rem & 31;

    // ---- stage 2: BN4, emit z_e_x (NCHW), A fragment, row norms ----
    v16h a;
    float part = 0.f;
    #pragma unroll
    for (int run = 0; run < 2; ++run) {
        int c0 = run * 16 + hs * 8;
        const float* p = &zstage[lr * 32 + c0];
        #pragma unroll
        for (int j = 0; j < 8; ++j) {
            int c = c0 + j;
            float z = fmaf(p[j], s_sc[c], s_sh[c]);   // no ReLU on layer 4
            zex[(((size_t)nimg * 32 + c) * 32 + h) * 32 + w] = z;
            part += z * z;
            a[run * 8 + j] = (_Float16)z;
        }
    }
    part += __shfl_xor(part, 16, 32);
    if (lane < 16) zn[wave * 16 + lane] = part;
    __syncthreads();

    float zrow[8];
    #pragma unroll
    for (int r = 0; r < 8; ++r) zrow[r] = zn[wave * 16 + r + hs * 8];

    float bv[8]; int bi[8];
    #pragma unroll
    for (int r = 0; r < 8; ++r) { bv[r] = 3.4e38f; bi[r] = 0; }

    // ---- stage 3: 512 codes = 32 N-tiles of one 16x16x32 WMMA (K = C = 32) ----
    // ncol ascends, so strict '<' keeps the lowest index on ties.
    for (int t = 0; t < 32; ++t) {
        int ncol = t * 16 + (lane & 15);
        const _Float16* bp = &E[ncol * 32 + hs * 8];
        v16h b;
        #pragma unroll
        for (int j = 0; j < 16; ++j) b[j] = bp[(j & 8) + j];
        v8f c0 = (v8f){};
        v8f dot = __builtin_amdgcn_wmma_f32_16x16x32_f16(
            false, a, false, b, (short)0, c0, false, false);
        float e2 = en[ncol];
        #pragma unroll
        for (int r = 0; r < 8; ++r) {
            float d = zrow[r] - 2.f * dot[r] + e2;
            if (d < bv[r]) { bv[r] = d; bi[r] = ncol; }
        }
    }
    // argmin across the 16 lanes sharing each C-tile row (tie -> lowest index)
    #pragma unroll
    for (int r = 0; r < 8; ++r) {
        float v = bv[r]; int i = bi[r];
        #pragma unroll
        for (int mask = 1; mask < 16; mask <<= 1) {
            float ov = __shfl_xor(v, mask, 32);
            int   oi = __shfl_xor(i, mask, 32);
            if (ov < v || (ov == v && oi < i)) { v = ov; i = oi; }
        }
        bv[r] = v; bi[r] = i;
    }
    if ((lane & 15) == 0) {
        #pragma unroll
        for (int r = 0; r < 8; ++r) {
            int row = r + hs * 8;
            best[wave * 16 + row] = bi[r];
            lat[m_base + row] = bi[r];
        }
    }
    __syncthreads();

    // ---- stage 4: z_q_x gather from f32 LDS codebook; lane <-> channel ----
    #pragma unroll
    for (int row = 0; row < 16; ++row) {
        int idx = best[wave * 16 + row];
        int mr = m_base + row;
        int ni = mr >> 10, rr = mr & 1023, hh = rr >> 5, ww = rr & 31;
        zqx[(((size_t)ni * 32 + lane) * 32 + hh) * 32 + ww] = Ef32[idx * 32 + lane];
    }
}

// ---------------------------------------------------------------
extern "C" void kernel_launch(void* const* d_in, const int* in_sizes, int n_in,
                              void* d_out, int out_size, void* d_ws, size_t ws_size,
                              hipStream_t stream) {
    (void)in_sizes; (void)n_in; (void)out_size; (void)ws_size;
    const float* x   = (const float*)d_in[0];
    const float* W1  = (const float*)d_in[1];
    const float* b1  = (const float*)d_in[2];
    const float* g1  = (const float*)d_in[3];
    const float* be1 = (const float*)d_in[4];
    const float* W2  = (const float*)d_in[5];
    const float* b2  = (const float*)d_in[6];
    const float* g2  = (const float*)d_in[7];
    const float* be2 = (const float*)d_in[8];
    const float* W3  = (const float*)d_in[9];
    const float* b3  = (const float*)d_in[10];
    const float* g3  = (const float*)d_in[11];
    const float* be3 = (const float*)d_in[12];
    const float* W4  = (const float*)d_in[13];
    const float* b4  = (const float*)d_in[14];
    const float* g4  = (const float*)d_in[15];
    const float* be4 = (const float*)d_in[16];
    const float* emb = (const float*)d_in[17];

    float* ws    = (float*)d_ws;
    float* stats = ws + WS_STATS;
    float* bn    = ws + WS_BN;
    float* y1    = ws + WS_Y1;
    float* y2    = ws + WS_Y2;
    float* y3    = ws + WS_Y3;
    float* zpre  = ws + WS_ZPRE;

    float* outf = (float*)d_out;
    float* zex  = outf;
    float* zqx  = outf + 4194304;
    int*   lat  = (int*)(outf + 8388608);

    k_zero_stats<<<1, 256, 0, stream>>>(stats);

    k_conv1<<<8192, 256, 0, stream>>>(x, W1, b1, y1, stats + 0);
    k_bnfold<<<1, 64, 0, stream>>>(stats + 0, g1, be1, bn + 0, 16, 1.f / 2097152.f);

    k_conv_wmma<16, 32, 128, 128, 64, 64, 4, 2, 1>
        <<<4096, 256, 0, stream>>>(y1, W2, b2, bn + 0, y2, stats + 128);
    k_bnfold<<<1, 64, 0, stream>>>(stats + 128, g2, be2, bn + 128, 32, 1.f / 524288.f);

    k_conv_wmma<32, 64, 64, 64, 32, 32, 4, 2, 1>
        <<<1024, 256, 0, stream>>>(y2, W3, b3, bn + 128, y3, stats + 256);
    k_bnfold<<<1, 64, 0, stream>>>(stats + 256, g3, be3, bn + 256, 64, 1.f / 131072.f);

    k_conv_wmma<64, 32, 32, 32, 32, 32, 1, 1, 0>
        <<<1024, 256, 0, stream>>>(y3, W4, b4, bn + 256, zpre, stats + 384);
    k_bnfold<<<1, 64, 0, stream>>>(stats + 384, g4, be4, bn + 384, 32, 1.f / 131072.f);

    k_vq<<<1024, 256, 0, stream>>>(zpre, emb, bn + 384, zex, zqx, lat);
}
